// CGCN_59193239273656
// MI455X (gfx1250) — matrix operand order repaired and checked by reading, not compiled, not run on hardware
//
#include <hip/hip_runtime.h>
#include <hip/hip_bf16.h>
#include <math.h>

#define NUM_USER   100000
#define NUM_ITEM   100000
#define N_NODES    200000
#define DIM_C      64
#define DIM_F      768
#define NUM_EDGES  1000000
#define NUM_ROUTING 3
#define NEG_SLOPE  0.01f

typedef _Float16 v16h __attribute__((ext_vector_type(16)));
typedef float    v8f  __attribute__((ext_vector_type(8)));

// ---------------------------------------------------------------------------
// helpers
// ---------------------------------------------------------------------------
__device__ __forceinline__ float reduce16(float s) {
    // butterfly reduce within each 16-lane half of the wave32
    #pragma unroll
    for (int off = 1; off < 16; off <<= 1) s += __shfl_xor(s, off, 32);
    return s;
}

__device__ __forceinline__ float leaky(float z) {
    return z > 0.0f ? z : NEG_SLOPE * z;
}

// ---------------------------------------------------------------------------
// 0) convert W [64,768] f32 -> f16 (stays hot in L2 for the GEMM)
// ---------------------------------------------------------------------------
__global__ void convW_kernel(const float* __restrict__ W, _Float16* __restrict__ Wh) {
    int i = blockIdx.x * blockDim.x + threadIdx.x;
    if (i < DIM_C * DIM_F) Wh[i] = (_Float16)W[i];
}

// ---------------------------------------------------------------------------
// 1) feats = l2norm(leaky_relu(feature @ W^T + b))  ->  x[NUM_USER .. N)
//    One wave computes a 16x64 output tile: 4 WMMA 16x16 accumulators,
//    K-loop over 768 in steps of 32 using v_wmma_f32_16x16x32_f16.
//    A layout (16-bit A 16x32, contiguous K as for B): row = lane&15,
//      lanes 0-15 hold K kk..kk+15; lanes 16-31 K kk+16..kk+31.
//    B layout (32x16): column = lane&15, lanes 0-15 K=kk..kk+15,
//      lanes 16-31 K=kk+16..kk+31 (16 contiguous f16 per lane).
// ---------------------------------------------------------------------------
__global__ void gemm_feats_kernel(const float* __restrict__ feature,
                                  const _Float16* __restrict__ Wh,
                                  const float* __restrict__ bias,
                                  float* __restrict__ x) {
    int gtid = blockIdx.x * blockDim.x + threadIdx.x;
    int wave = gtid >> 5;
    int lane = threadIdx.x & 31;
    int row0 = wave * 16;
    if (row0 >= NUM_ITEM) return;              // wave-uniform: EXEC stays all-1s

    int m  = lane & 15;                        // A row / B column within tile
    int hi = lane >> 4;                        // K-half select

    v8f acc0 = {}, acc1 = {}, acc2 = {}, acc3 = {};

    const float* arow = feature + (size_t)(row0 + m) * DIM_F;

    for (int kk = 0; kk < DIM_F; kk += 32) {
        // ---- A tile: 16 f16 per lane, 16 contiguous floats ----
        const float* ap = arow + kk + hi * 16;
        __builtin_prefetch(ap + 32, 0, 1);     // next K-iter chunk -> global_prefetch
        v16h a;
        #pragma unroll
        for (int i = 0; i < 8; ++i) a[i]     = (_Float16)ap[i];
        #pragma unroll
        for (int i = 0; i < 8; ++i) a[8 + i] = (_Float16)ap[8 + i];

        // ---- B tiles: 16 contiguous f16 per lane from W row (= out channel) ----
        const _Float16* bbase = Wh + kk + hi * 16;
        v16h b0 = *(const v16h*)(bbase + (size_t)(0 * 16 + m) * DIM_F);
        v16h b1 = *(const v16h*)(bbase + (size_t)(1 * 16 + m) * DIM_F);
        v16h b2 = *(const v16h*)(bbase + (size_t)(2 * 16 + m) * DIM_F);
        v16h b3 = *(const v16h*)(bbase + (size_t)(3 * 16 + m) * DIM_F);

        acc0 = __builtin_amdgcn_wmma_f32_16x16x32_f16(false, a, false, b0, (short)0, acc0, false, false);
        acc1 = __builtin_amdgcn_wmma_f32_16x16x32_f16(false, a, false, b1, (short)0, acc1, false, false);
        acc2 = __builtin_amdgcn_wmma_f32_16x16x32_f16(false, a, false, b2, (short)0, acc2, false, false);
        acc3 = __builtin_amdgcn_wmma_f32_16x16x32_f16(false, a, false, b3, (short)0, acc3, false, false);
    }

    // C layout: VGPR r of lanes 0-15 holds row r, lanes 16-31 row r+8; col = t*16+m.
    float b0s = bias[0 * 16 + m];
    float b1s = bias[1 * 16 + m];
    float b2s = bias[2 * 16 + m];
    float b3s = bias[3 * 16 + m];

    #pragma unroll
    for (int r = 0; r < 8; ++r) {
        float v0 = leaky(acc0[r] + b0s);
        float v1 = leaky(acc1[r] + b1s);
        float v2 = leaky(acc2[r] + b2s);
        float v3 = leaky(acc3[r] + b3s);
        float s  = reduce16(v0 * v0 + v1 * v1 + v2 * v2 + v3 * v3);
        float inv = 1.0f / fmaxf(sqrtf(s), 1e-12f);
        int orow = row0 + r + hi * 8;
        float* xp = x + (size_t)(NUM_USER + orow) * DIM_C;
        xp[0 * 16 + m] = v0 * inv;
        xp[1 * 16 + m] = v1 * inv;
        xp[2 * 16 + m] = v2 * inv;
        xp[3 * 16 + m] = v3 * inv;
    }
}

// ---------------------------------------------------------------------------
// 2) x[0..NUM_USER) = l2norm(preference)   (16 lanes per row, float4)
// ---------------------------------------------------------------------------
__global__ void l2norm_rows_kernel(const float* __restrict__ in,
                                   float* __restrict__ out, int nrows) {
    int t = blockIdx.x * blockDim.x + threadIdx.x;
    int row = t >> 4, sub = t & 15;
    if (row >= nrows) return;
    float4 v = ((const float4*)(in + (size_t)row * DIM_C))[sub];
    float s = reduce16(v.x * v.x + v.y * v.y + v.z * v.z + v.w * v.w);
    float inv = 1.0f / fmaxf(sqrtf(s), 1e-12f);
    float4 o = {v.x * inv, v.y * inv, v.z * inv, v.w * inv};
    ((float4*)(out + (size_t)row * DIM_C))[sub] = o;
}

// ---------------------------------------------------------------------------
// 3) per edge: e = exp(dot(x[dst], x[src])) ; denom[dst] += e
//    (rows are unit-norm => dot in [-1,1]; segment-max is a softmax no-op)
//    e >= NUM_EDGES means reversed edge (final doubled pass).
// ---------------------------------------------------------------------------
__global__ void edge_attn_kernel(const float* __restrict__ x,
                                 const int* __restrict__ src,
                                 const int* __restrict__ dst,
                                 float* __restrict__ evals,
                                 float* __restrict__ denom, int ne_total) {
    int t = blockIdx.x * blockDim.x + threadIdx.x;
    int e = t >> 4, sub = t & 15;
    if (e >= ne_total) return;
    int s_, d_;
    if (e < NUM_EDGES) { s_ = src[e]; d_ = dst[e]; }
    else               { s_ = dst[e - NUM_EDGES]; d_ = src[e - NUM_EDGES]; }
    float4 a = ((const float4*)(x + (size_t)s_ * DIM_C))[sub];
    float4 c = ((const float4*)(x + (size_t)d_ * DIM_C))[sub];
    float dot = reduce16(a.x * c.x + a.y * c.y + a.z * c.z + a.w * c.w);
    if (sub == 0) {
        float ev = expf(dot);
        evals[e] = ev;
        atomicAdd(denom + d_, ev);
    }
}

// ---------------------------------------------------------------------------
// 4) per edge: alpha = e/(denom+eps) ; h[dst] += alpha * x[src] (atomics)
// ---------------------------------------------------------------------------
__global__ void edge_scatter_kernel(const float* __restrict__ x,
                                    const int* __restrict__ src,
                                    const int* __restrict__ dst,
                                    const float* __restrict__ evals,
                                    const float* __restrict__ denom,
                                    float* __restrict__ h,
                                    float* __restrict__ alpha_out, int ne_total) {
    int t = blockIdx.x * blockDim.x + threadIdx.x;
    int e = t >> 4, sub = t & 15;
    if (e >= ne_total) return;
    int s_, d_;
    if (e < NUM_EDGES) { s_ = src[e]; d_ = dst[e]; }
    else               { s_ = dst[e - NUM_EDGES]; d_ = src[e - NUM_EDGES]; }
    float a = evals[e] / (denom[d_] + 1e-16f);
    if (alpha_out && sub == 0) alpha_out[e] = a;
    float4 xj = ((const float4*)(x + (size_t)s_ * DIM_C))[sub];
    float* hp = h + (size_t)d_ * DIM_C + sub * 4;
    atomicAdd(hp + 0, xj.x * a);
    atomicAdd(hp + 1, xj.y * a);
    atomicAdd(hp + 2, xj.z * a);
    atomicAdd(hp + 3, xj.w * a);
}

// ---------------------------------------------------------------------------
// 5) preference update (in place in x): x_u = l2norm(x_u + h_u)
// ---------------------------------------------------------------------------
__global__ void update_pref_kernel(float* __restrict__ x, const float* __restrict__ h) {
    int t = blockIdx.x * blockDim.x + threadIdx.x;
    int row = t >> 4, sub = t & 15;
    if (row >= NUM_USER) return;
    float4* xp = (float4*)(x + (size_t)row * DIM_C);
    float4 v  = xp[sub];
    float4 hv = ((const float4*)(h + (size_t)row * DIM_C))[sub];
    v.x += hv.x; v.y += hv.y; v.z += hv.z; v.w += hv.w;
    float s = reduce16(v.x * v.x + v.y * v.y + v.z * v.z + v.w * v.w);
    float inv = 1.0f / fmaxf(sqrtf(s), 1e-12f);
    v.x *= inv; v.y *= inv; v.z *= inv; v.w *= inv;
    xp[sub] = v;
}

// ---------------------------------------------------------------------------
// 6) out = x + leaky_relu(h)
// ---------------------------------------------------------------------------
__global__ void finalize_kernel(const float* __restrict__ x,
                                const float* __restrict__ h,
                                float* __restrict__ out) {
    int t = blockIdx.x * blockDim.x + threadIdx.x;
    if (t >= N_NODES * (DIM_C / 4)) return;
    float4 xv = ((const float4*)x)[t];
    float4 hv = ((const float4*)h)[t];
    float4 o;
    o.x = xv.x + leaky(hv.x);
    o.y = xv.y + leaky(hv.y);
    o.z = xv.z + leaky(hv.z);
    o.w = xv.w + leaky(hv.w);
    ((float4*)out)[t] = o;
}

// ---------------------------------------------------------------------------
extern "C" void kernel_launch(void* const* d_in, const int* in_sizes, int n_in,
                              void* d_out, int out_size, void* d_ws, size_t ws_size,
                              hipStream_t stream) {
    const float* feature = (const float*)d_in[0];   // [NUM_ITEM, 768]
    const float* pref    = (const float*)d_in[1];   // [NUM_USER, 64]
    const float* W       = (const float*)d_in[2];   // [64, 768]
    const float* bias    = (const float*)d_in[3];   // [64]
    const int*   ei      = (const int*)d_in[4];     // [2, E]
    const int* src = ei;
    const int* dst = ei + NUM_EDGES;
    float* out = (float*)d_out;

    char* ws = (char*)d_ws;
    size_t off = 0;
    _Float16* Wh   = (_Float16*)(ws + off); off += (size_t)DIM_C * DIM_F * 2;   // 96 KB
    float*    x    = (float*)(ws + off);    off += (size_t)N_NODES * DIM_C * 4; // 51.2 MB
    float*    h    = (float*)(ws + off);    off += (size_t)N_NODES * DIM_C * 4; // 51.2 MB
    float*    den  = (float*)(ws + off);    off += (size_t)N_NODES * 4;         // 0.8 MB
    float*    ev   = (float*)(ws + off);    off += (size_t)2 * NUM_EDGES * 4;   // 8 MB
    if (off > ws_size) return;

    // feats via WMMA GEMM (fused bias + leaky_relu + l2norm), pref l2norm
    convW_kernel<<<(DIM_C * DIM_F + 255) / 256, 256, 0, stream>>>(W, Wh);
    {
        int threads = (NUM_ITEM / 16) * 32;     // one wave per 16 rows
        gemm_feats_kernel<<<(threads + 255) / 256, 256, 0, stream>>>(feature, Wh, bias, x);
    }
    l2norm_rows_kernel<<<(NUM_USER * 16 + 255) / 256, 256, 0, stream>>>(pref, x, NUM_USER);

    // 3 routing iterations over the bipartite edges
    for (int it = 0; it < NUM_ROUTING; ++it) {
        hipMemsetAsync(den, 0, (size_t)N_NODES * 4, stream);
        hipMemsetAsync(h,   0, (size_t)NUM_USER * DIM_C * 4, stream);
        edge_attn_kernel<<<(NUM_EDGES * 16 + 255) / 256, 256, 0, stream>>>(
            x, src, dst, ev, den, NUM_EDGES);
        edge_scatter_kernel<<<(NUM_EDGES * 16 + 255) / 256, 256, 0, stream>>>(
            x, src, dst, ev, den, h, nullptr, NUM_EDGES);
        update_pref_kernel<<<(NUM_USER * 16 + 255) / 256, 256, 0, stream>>>(x, h);
    }

    // final pass over doubled (bidirectional) edges; alpha -> tail of d_out
    hipMemsetAsync(den, 0, (size_t)N_NODES * 4, stream);
    hipMemsetAsync(h,   0, (size_t)N_NODES * DIM_C * 4, stream);
    edge_attn_kernel<<<(2 * NUM_EDGES * 16 + 255) / 256, 256, 0, stream>>>(
        x, src, dst, ev, den, 2 * NUM_EDGES);
    edge_scatter_kernel<<<(2 * NUM_EDGES * 16 + 255) / 256, 256, 0, stream>>>(
        x, src, dst, ev, den, h, out + (size_t)N_NODES * DIM_C, 2 * NUM_EDGES);
    finalize_kernel<<<(N_NODES * 16 + 255) / 256, 256, 0, stream>>>(x, h, out);
}
